// TemporalAtomAccumulator_86732569575516
// MI455X (gfx1250) — matrix operand order repaired and checked by reading, not compiled, not run on hardware
//
#include <hip/hip_runtime.h>
#include <hip/hip_bf16.h>

typedef __attribute__((ext_vector_type(16))) __bf16 v16bf;
typedef __attribute__((ext_vector_type(8)))  __bf16 v8bf;
typedef __attribute__((ext_vector_type(8)))  float  v8f;

#define B_  4
#define T_  4096
#define N_  256
#define D_  64
#define SCALE_ 0.125f   // D^-0.5 = 64^-0.5

__device__ __forceinline__ v8f zero_v8f() {
  v8f z;
#pragma unroll
  for (int i = 0; i < 8; ++i) z[i] = 0.0f;
  return z;
}

// ---------------------------------------------------------------------------
// Kernel 0: f32 -> bf16 conversion for the sequence (= V) and Wq/Wk.
// ---------------------------------------------------------------------------
__global__ void cvt_kernel(const float* __restrict__ X,
                           const float* __restrict__ Wq,
                           const float* __restrict__ Wk,
                           __bf16* __restrict__ Xb,
                           __bf16* __restrict__ Wqb,
                           __bf16* __restrict__ Wkb) {
  const int nchunk = (B_ * T_ * N_) / 8;
  int tid = blockIdx.x * blockDim.x + threadIdx.x;
  int stride = gridDim.x * blockDim.x;
  for (int c = tid; c < nchunk; c += stride) {
    const float* p = X + (size_t)c * 8;
    v8bf o;
#pragma unroll
    for (int e = 0; e < 8; ++e) o[e] = (__bf16)p[e];
    *(v8bf*)(Xb + (size_t)c * 8) = o;
  }
  for (int i = tid; i < D_ * N_; i += stride) {
    Wqb[i] = (__bf16)Wq[i];
    Wkb[i] = (__bf16)Wk[i];
  }
}

// ---------------------------------------------------------------------------
// Kernel 1: Q/K projections.  Grid = (B*T)/64 blocks, 128 threads (4 waves).
// Wave w computes 16 rows of Q and K (16x64 each) = 64 WMMAs.
// Q is pre-scaled by SCALE before bf16 conversion.
// ---------------------------------------------------------------------------
__global__ void __launch_bounds__(128)
proj_kernel(const __bf16* __restrict__ Xb,   // [B*T, N]
            const __bf16* __restrict__ Wqb,  // [D, N]
            const __bf16* __restrict__ Wkb,  // [D, N]
            const float*  __restrict__ bq,
            const float*  __restrict__ bk,
            __bf16* __restrict__ Qb,         // [B*T, D]
            __bf16* __restrict__ Kb) {       // [B*T, D]
  const int wv   = threadIdx.x >> 5;
  const int lane = threadIdx.x & 31;
  const int g    = lane >> 4;
  const int lm   = lane & 15;
  const int row0 = blockIdx.x * 64 + wv * 16;
  const int rowA = row0 + lm;

  v8f accQ[4], accK[4];
#pragma unroll
  for (int dt = 0; dt < 4; ++dt) { accQ[dt] = zero_v8f(); accK[dt] = zero_v8f(); }

#pragma unroll
  for (int ks = 0; ks < 8; ++ks) {
    const int kbase = ks * 32;
    // A fragment: X row rowA, K chunks {16g..16g+7} and {16g+8..16g+15}
    v16bf a;
    {
      const __bf16* p = Xb + (size_t)rowA * N_ + kbase + 16 * g;
      v8bf lo = *(const v8bf*)p;
      v8bf hi = *(const v8bf*)(p + 8);
#pragma unroll
      for (int e = 0; e < 8; ++e) { a[e] = lo[e]; a[e + 8] = hi[e]; }
    }
#pragma unroll
    for (int dt = 0; dt < 4; ++dt) {
      const int dcol = dt * 16 + lm;
      v16bf bf = *(const v16bf*)(Wqb + (size_t)dcol * N_ + kbase + 16 * g);
      accQ[dt] = __builtin_amdgcn_wmma_f32_16x16x32_bf16(
          false, a, false, bf, (short)0, accQ[dt], false, false);
      v16bf bf2 = *(const v16bf*)(Wkb + (size_t)dcol * N_ + kbase + 16 * g);
      accK[dt] = __builtin_amdgcn_wmma_f32_16x16x32_bf16(
          false, a, false, bf2, (short)0, accK[dt], false, false);
    }
  }

#pragma unroll
  for (int dt = 0; dt < 4; ++dt) {
    const int d = dt * 16 + lm;
    const float biasq = bq[d];
    const float biask = bk[d];
#pragma unroll
    for (int i = 0; i < 8; ++i) {
      const int row = row0 + i + 8 * g;
      Qb[(size_t)row * D_ + d] = (__bf16)((accQ[dt][i] + biasq) * SCALE_);
      Kb[(size_t)row * D_ + d] = (__bf16)(accK[dt][i] + biask);
    }
  }
}

// ---------------------------------------------------------------------------
// Kernel 2: causal flash attention.  Grid = (T/64, B), 128 threads (4 waves).
// Block owns 64 queries; wave w owns 16.  Loops over 64-key tiles <= diag.
// Next iteration's K/V tiles are prefetched (global_prefetch_b8) so staging
// loads at the top of the next iteration hit WGP$/L2.
// ---------------------------------------------------------------------------
constexpr int VT_STRIDE = 80;  // bf16 elems; 160B rows -> 32B aligned frags
constexpr int P_STRIDE  = 72;  // bf16 elems; 144B rows -> 16B aligned chunks

__global__ void __launch_bounds__(128)
flash_kernel(const __bf16* __restrict__ Qb,
             const __bf16* __restrict__ Kb,
             const __bf16* __restrict__ Vb,
             const int*    __restrict__ traj,
             float*        __restrict__ out) {
  __shared__ __bf16 Vt[N_ * VT_STRIDE];      // V transposed: [n][key]
  __shared__ __bf16 Pb[4 * 16 * P_STRIDE];   // per-wave P staging

  const int b    = blockIdx.y;
  const int qt   = blockIdx.x;
  const int tid  = threadIdx.x;
  const int wv   = tid >> 5;
  const int lane = tid & 31;
  const int g    = lane >> 4;
  const int lm   = lane & 15;
  const int q0   = qt * 64;

  // Q A-fragments for this wave's 16 rows (already scaled by SCALE)
  v16bf qa[2];
  {
    const __bf16* qrow = Qb + ((size_t)b * T_ + q0 + wv * 16 + lm) * D_;
#pragma unroll
    for (int ks = 0; ks < 2; ++ks) {
      v8bf lo = *(const v8bf*)(qrow + ks * 32 + 16 * g);
      v8bf hi = *(const v8bf*)(qrow + ks * 32 + 16 * g + 8);
#pragma unroll
      for (int e = 0; e < 8; ++e) { qa[ks][e] = lo[e]; qa[ks][e + 8] = hi[e]; }
    }
  }

  v8f acc[16];
#pragma unroll
  for (int nt = 0; nt < 16; ++nt) acc[nt] = zero_v8f();
  float m[8], s[8];
#pragma unroll
  for (int i = 0; i < 8; ++i) { m[i] = -1e30f; s[i] = 0.0f; }

  __bf16* pbase = Pb + wv * 16 * P_STRIDE;

  for (int j = 0; j <= qt; ++j) {
    __syncthreads();  // previous iteration's Vt readers are done
    // Cooperative transpose-load of V tile [64 keys][256 n] -> Vt[n][key]
    const __bf16* vsrc = Vb + ((size_t)b * T_ + j * 64) * N_;
#pragma unroll
    for (int it = 0; it < 16; ++it) {
      int c   = tid + it * 128;   // 2048 chunks of 8 elems
      int key = c >> 5;
      int nc  = (c & 31) * 8;
      v8bf v = *(const v8bf*)(vsrc + (size_t)key * N_ + nc);
#pragma unroll
      for (int e = 0; e < 8; ++e) Vt[(nc + e) * VT_STRIDE + key] = v[e];
    }
    __syncthreads();

    // S = Q K^T : 16 rows x 64 keys = 4 tiles x 2 k-steps
    v8f st[4];
#pragma unroll
    for (int nt = 0; nt < 4; ++nt) st[nt] = zero_v8f();
#pragma unroll
    for (int nt = 0; nt < 4; ++nt) {
      const __bf16* krow = Kb + ((size_t)b * T_ + j * 64 + nt * 16 + lm) * D_;
#pragma unroll
      for (int ks = 0; ks < 2; ++ks) {
        v16bf kf = *(const v16bf*)(krow + ks * 32 + 16 * g);
        st[nt] = __builtin_amdgcn_wmma_f32_16x16x32_bf16(
            false, qa[ks], false, kf, (short)0, st[nt], false, false);
      }
    }

    // Prefetch next iteration's V and K tiles into cache (global_prefetch_b8).
    // Block-uniform branch: EXEC stays all-ones for the WMMAs.
    if (j < qt) {
      const char* nv = (const char*)(Vb + ((size_t)b * T_ + (j + 1) * 64) * N_);
      __builtin_prefetch(nv + (size_t)tid * 128, 0, 3);           // 16KB
      __builtin_prefetch(nv + 16384 + (size_t)tid * 128, 0, 3);   // 16KB
      const char* nk = (const char*)(Kb + ((size_t)b * T_ + (j + 1) * 64) * D_);
      __builtin_prefetch(nk + (size_t)tid * 64, 0, 3);            // 8KB
    }

    // Causal mask on the diagonal tile
    if (j == qt) {
#pragma unroll
      for (int nt = 0; nt < 4; ++nt) {
        const int col = nt * 16 + lm;
#pragma unroll
        for (int i = 0; i < 8; ++i) {
          const int row = wv * 16 + i + 8 * g;
          if (col > row) st[nt][i] = -1e30f;
        }
      }
    }

    // Online softmax: rows live in 16-lane groups -> xor-shuffle reductions
    float nm[8], rs[8];
#pragma unroll
    for (int i = 0; i < 8; ++i) {
      float v = fmaxf(fmaxf(st[0][i], st[1][i]), fmaxf(st[2][i], st[3][i]));
      v = fmaxf(v, __shfl_xor(v, 1, 32));
      v = fmaxf(v, __shfl_xor(v, 2, 32));
      v = fmaxf(v, __shfl_xor(v, 4, 32));
      v = fmaxf(v, __shfl_xor(v, 8, 32));
      nm[i] = fmaxf(v, m[i]);
    }
#pragma unroll
    for (int i = 0; i < 8; ++i) {
      float r = 0.0f;
#pragma unroll
      for (int nt = 0; nt < 4; ++nt) {
        float p = __expf(st[nt][i] - nm[i]);
        st[nt][i] = p;
        r += p;
      }
      r += __shfl_xor(r, 1, 32);
      r += __shfl_xor(r, 2, 32);
      r += __shfl_xor(r, 4, 32);
      r += __shfl_xor(r, 8, 32);
      rs[i] = r;
    }
#pragma unroll
    for (int i = 0; i < 8; ++i) {
      const float corr = __expf(m[i] - nm[i]);
      s[i] = s[i] * corr + rs[i];
      m[i] = nm[i];
#pragma unroll
      for (int nt = 0; nt < 16; ++nt) acc[nt][i] *= corr;
    }

    // Stage P (C layout) -> row-major LDS -> reload as A fragments
#pragma unroll
    for (int nt = 0; nt < 4; ++nt)
#pragma unroll
      for (int i = 0; i < 8; ++i)
        pbase[(i + 8 * g) * P_STRIDE + nt * 16 + lm] = (__bf16)st[nt][i];

    v16bf pa[2];
#pragma unroll
    for (int ks = 0; ks < 2; ++ks) {
      const __bf16* prow = pbase + lm * P_STRIDE + ks * 32 + 16 * g;
      v8bf lo = *(const v8bf*)prow;
      v8bf hi = *(const v8bf*)(prow + 8);
#pragma unroll
      for (int e = 0; e < 8; ++e) { pa[ks][e] = lo[e]; pa[ks][e + 8] = hi[e]; }
    }

    // O += P @ V : 16 n-tiles x 2 k-steps = 32 WMMAs
#pragma unroll
    for (int nt = 0; nt < 16; ++nt) {
#pragma unroll
      for (int ks = 0; ks < 2; ++ks) {
        v16bf vf = *(const v16bf*)(&Vt[(nt * 16 + lm) * VT_STRIDE + ks * 32 + 16 * g]);
        acc[nt] = __builtin_amdgcn_wmma_f32_16x16x32_bf16(
            false, pa[ks], false, vf, (short)0, acc[nt], false, false);
      }
    }
  }

  // Epilogue: normalize, traj-length mask, write f32 output
  const int tl = traj[b];
  float inv[8];
#pragma unroll
  for (int i = 0; i < 8; ++i) inv[i] = 1.0f / s[i];
#pragma unroll
  for (int nt = 0; nt < 16; ++nt) {
#pragma unroll
    for (int i = 0; i < 8; ++i) {
      const int row = q0 + wv * 16 + i + 8 * g;
      float val = acc[nt][i] * inv[i];
      if (row >= tl) val = 0.0f;
      out[((size_t)b * T_ + row) * N_ + nt * 16 + lm] = val;
    }
  }
}

// ---------------------------------------------------------------------------
extern "C" void kernel_launch(void* const* d_in, const int* in_sizes, int n_in,
                              void* d_out, int out_size, void* d_ws, size_t ws_size,
                              hipStream_t stream) {
  const float* X    = (const float*)d_in[0];  // [B,T,N]
  const float* Wq   = (const float*)d_in[1];  // [D,N]
  const float* bq   = (const float*)d_in[2];  // [D]
  const float* Wk   = (const float*)d_in[3];  // [D,N]
  const float* bk   = (const float*)d_in[4];  // [D]
  const int*   traj = (const int*)d_in[5];    // [B]
  float* out = (float*)d_out;

  // Workspace layout (bf16): Xb | Qb | Kb | Wqb | Wkb  (~12 MiB total)
  __bf16* Xb  = (__bf16*)d_ws;
  __bf16* Qb  = Xb  + (size_t)B_ * T_ * N_;
  __bf16* Kb  = Qb  + (size_t)B_ * T_ * D_;
  __bf16* Wqb = Kb  + (size_t)B_ * T_ * D_;
  __bf16* Wkb = Wqb + (size_t)D_ * N_;

  cvt_kernel<<<2048, 256, 0, stream>>>(X, Wq, Wk, Xb, Wqb, Wkb);
  proj_kernel<<<(B_ * T_) / 64, 128, 0, stream>>>(Xb, Wqb, Wkb, bq, bk, Qb, Kb);
  flash_kernel<<<dim3(T_ / 64, B_), 128, 0, stream>>>(Qb, Kb, Xb, traj, out);
}
